// QuantumInspiredAttention_5703716569513
// MI455X (gfx1250) — matrix operand order repaired and checked by reading, not compiled, not run on hardware
//
#include <hip/hip_runtime.h>
#include <hip/hip_bf16.h>

typedef __attribute__((ext_vector_type(16))) _Float16 v16h;
typedef __attribute__((ext_vector_type(8)))  _Float16 v8h;
typedef __attribute__((ext_vector_type(2)))  __fp16   fp16x2;
typedef __attribute__((ext_vector_type(8)))  float    v8f;
typedef __attribute__((ext_vector_type(4)))  float    v4f;

#define NROWS 2048      // BATCH*SEQ
#define DIM   512
#define NHEAD 16384     // BATCH*SEQ*HEADS
#define DH    64

// ---------------------------------------------------------------------------
// Kernel 1: rank-4 TT projections q/k/v, per-head L2-normalize (embed),
// fold 1/sqrt(64) into q, emit f16 qh/kh [16384,64] and transposed vT [64,16384]
// ---------------------------------------------------------------------------
__global__ __launch_bounds__(256) void qia_proj_kernel(
    const float* __restrict__ x,
    const float* __restrict__ qA, const float* __restrict__ qB, const float* __restrict__ qb,
    const float* __restrict__ kA, const float* __restrict__ kB, const float* __restrict__ kb_,
    const float* __restrict__ vA, const float* __restrict__ vB, const float* __restrict__ vb,
    _Float16* __restrict__ qh, _Float16* __restrict__ kh, _Float16* __restrict__ vT)
{
    __shared__ float xs[DIM];
    __shared__ float ybuf[3][DIM];
    __shared__ float red[8 * 12];
    __shared__ float tv[12];
    __shared__ float nrm[16];

    const int row = blockIdx.x;          // 0..2047
    const int tid = threadIdx.x;
    const float* xr = x + row * DIM;
    xs[tid]       = xr[tid];
    xs[tid + 256] = xr[tid + 256];
    if (tid < 16) nrm[tid] = 0.f;
    __syncthreads();

    // t[r] = sum_d x[d] * B[r][d]   for q,k,v  (12 values)
    float acc[12];
#pragma unroll
    for (int i = 0; i < 12; i++) acc[i] = 0.f;
    for (int d = tid; d < DIM; d += 256) {
        float xv = xs[d];
#pragma unroll
        for (int r = 0; r < 4; r++) {
            acc[r]     += xv * qB[r * DIM + d];
            acc[4 + r] += xv * kB[r * DIM + d];
            acc[8 + r] += xv * vB[r * DIM + d];
        }
    }
#pragma unroll
    for (int i = 0; i < 12; i++) {
        float v = acc[i];
#pragma unroll
        for (int off = 16; off; off >>= 1) v += __shfl_xor(v, off, 32);
        if ((tid & 31) == 0) red[(tid >> 5) * 12 + i] = v;
    }
    __syncthreads();
    if (tid < 12) {
        float s = 0.f;
#pragma unroll
        for (int w = 0; w < 8; w++) s += red[w * 12 + tid];
        tv[tid] = s;
    }
    __syncthreads();

    // y[o] = b[o] + sum_r t[r]*A[o][r] ; accumulate per-64-chunk norms for q,k
    for (int o = tid; o < DIM; o += 256) {
        float yq = qb[o], yk = kb_[o], yv = vb[o];
#pragma unroll
        for (int r = 0; r < 4; r++) {
            yq += tv[r]     * qA[o * 4 + r];
            yk += tv[4 + r] * kA[o * 4 + r];
            yv += tv[8 + r] * vA[o * 4 + r];
        }
        ybuf[0][o] = yq; ybuf[1][o] = yk; ybuf[2][o] = yv;
        int ch = o >> 6;
        atomicAdd(&nrm[ch],     yq * yq);
        atomicAdd(&nrm[8 + ch], yk * yk);
    }
    __syncthreads();

    for (int o = tid; o < DIM; o += 256) {
        int ch = o >> 6;
        float inq = 0.125f / (sqrtf(nrm[ch])     + 1e-8f);  // 1/sqrt(DH) folded in
        float ink = 1.0f   / (sqrtf(nrm[8 + ch]) + 1e-8f);
        int gi = row * DIM + o;
        qh[gi] = (_Float16)(ybuf[0][o] * inq);
        kh[gi] = (_Float16)(ybuf[1][o] * ink);
        int n = row * 8 + (o >> 6);
        int d = o & 63;
        vT[(size_t)d * NHEAD + n] = (_Float16)ybuf[2][o];
    }
}

// ---------------------------------------------------------------------------
// Kernel 2: flash attention over N=16384 rows, dh=64, transposed formulation.
// One wave per 16-row q tile. S^T = K Q^T, O^T += V^T P^T via WMMA f16.
// Half-wave (xor-16) exchanges use v_permlanex16_b32 (VALU) instead of LDS.
// ---------------------------------------------------------------------------
static __device__ inline v16h cat8(v8h a, v8h b) {
    v16h r;
#pragma unroll
    for (int i = 0; i < 8; i++) { r[i] = a[i]; r[8 + i] = b[i]; }
    return r;
}

static __device__ inline v8f wmma_f16(v16h a, v16h b, v8f c) {
    return __builtin_amdgcn_wmma_f32_16x16x32_f16(false, a, false, b,
                                                  (short)0, c, false, false);
}

static __device__ inline unsigned pk2(float a, float b) {
    union { fp16x2 h; unsigned u; } cv;
    cv.h = __builtin_amdgcn_cvt_pkrtz(a, b);
    return cv.u;
}

// identity swap of lane halves 0-15 <-> 16-31 (pure VALU, no LDS)
static __device__ inline int swap16i(int v) {
    return __builtin_amdgcn_permlanex16(v, v, 0x76543210, 0xfedcba98, false, false);
}
static __device__ inline float swap16f(float v) {
    union { float f; int i; } a, b;
    a.f = v;
    b.i = swap16i(a.i);
    return b.f;
}

__global__ __launch_bounds__(256) void qia_attn_kernel(
    const _Float16* __restrict__ qh, const _Float16* __restrict__ kh,
    const _Float16* __restrict__ vT, float* __restrict__ att)
{
    const int wave  = threadIdx.x >> 5;
    const int lane  = threadIdx.x & 31;
    const int qt    = blockIdx.x * 8 + wave;   // 0..1023
    const int qbase = qt * 16;
    const int nlo   = lane & 15;
    const int hi    = lane >> 4;               // 0 or 1
    const bool low  = (hi == 0);

    // Q in B-operand layout: lane n holds Q[qbase+n][8*hi + 0..7, 16 + 8*hi + 0..7] etc.
    const _Float16* qrow = qh + (size_t)(qbase + nlo) * DH;
    const v16h bq0 = cat8(*(const v8h*)(qrow + hi * 8), *(const v8h*)(qrow + 16 + hi * 8));        // dh 0..31
    const v16h bq1 = cat8(*(const v8h*)(qrow + 32 + hi * 8), *(const v8h*)(qrow + 48 + hi * 8));   // dh 32..63

    v8f oacc[4];
#pragma unroll
    for (int s = 0; s < 4; s++) oacc[s] = (v8f){};
    float m_run = -1e30f, l_run = 0.f;

    for (int kb = 0; kb < NHEAD; kb += 32) {
        // ---- S^T = K_tile(32x64) * Q^T(64x16): two 16x16 C tiles ----
        const _Float16* k0r = kh + (size_t)(kb + nlo) * DH;
        const _Float16* k1r = k0r + 16 * DH;
        v16h ak;
        v8f t0 = (v8f){}, t1 = (v8f){};
        ak = cat8(*(const v8h*)(k0r + hi * 8), *(const v8h*)(k0r + 16 + hi * 8));
        t0 = wmma_f16(ak, bq0, t0);
        ak = cat8(*(const v8h*)(k0r + 32 + hi * 8), *(const v8h*)(k0r + 48 + hi * 8));
        t0 = wmma_f16(ak, bq1, t0);
        ak = cat8(*(const v8h*)(k1r + hi * 8), *(const v8h*)(k1r + 16 + hi * 8));
        t1 = wmma_f16(ak, bq0, t1);
        ak = cat8(*(const v8h*)(k1r + 32 + hi * 8), *(const v8h*)(k1r + 48 + hi * 8));
        t1 = wmma_f16(ak, bq1, t1);

        // ---- online softmax stats (per q-column = per lane pair) ----
        float smax = t0[0];
#pragma unroll
        for (int r = 1; r < 8; r++) smax = fmaxf(smax, t0[r]);
#pragma unroll
        for (int r = 0; r < 8; r++) smax = fmaxf(smax, t1[r]);
        smax = fmaxf(smax, swap16f(smax));
        float mnew  = fmaxf(m_run, smax);
        float scale = __expf(m_run - mnew);
        float rs = 0.f;
#pragma unroll
        for (int r = 0; r < 8; r++) { t0[r] = __expf(t0[r] - mnew); rs += t0[r]; }
#pragma unroll
        for (int r = 0; r < 8; r++) { t1[r] = __expf(t1[r] - mnew); rs += t1[r]; }
        rs += swap16f(rs);
        l_run = l_run * scale + rs;
        m_run = mnew;
#pragma unroll
        for (int s = 0; s < 4; s++)
#pragma unroll
            for (int r = 0; r < 8; r++) oacc[s][r] *= scale;

        // ---- build P^T B-operand (32k x 16q) from C-layout tiles ----
        // B layout: lanes 0-15 want col n, k0..15; lanes 16-31 want col n, k16..31.
        // Own half supplies k{0..7|24..31}; partner half supplies the middle.
        unsigned t0u[4], t1u[4];
#pragma unroll
        for (int j = 0; j < 4; j++) {
            t0u[j] = pk2(t0[2 * j], t0[2 * j + 1]);
            t1u[j] = pk2(t1[2 * j], t1[2 * j + 1]);
        }
        union { v16h h; unsigned u[8]; } bp;
#pragma unroll
        for (int j = 0; j < 4; j++) {
            // give the partner half what IT needs, then swap once
            unsigned z  = low ? t1u[j] : t0u[j];
            unsigned zx = (unsigned)swap16i((int)z);
            bp.u[j]     = low ? t0u[j] : zx;    // k 0..7   | k 16..23
            bp.u[4 + j] = low ? zx     : t1u[j];// k 8..15  | k 24..31
        }

        // ---- O^T += V^T(16dh x 32n) * P^T(32n x 16q), 4 dh slices ----
#pragma unroll
        for (int s = 0; s < 4; s++) {
            const _Float16* vr = vT + (size_t)(s * 16 + nlo) * NHEAD + kb;
            v16h av = cat8(*(const v8h*)(vr + hi * 16), *(const v8h*)(vr + hi * 16 + 8));
            oacc[s] = wmma_f16(av, bp.h, oacc[s]);
        }
    }

    // ---- epilogue: softmax denom, then _measure (squared renormalize) ----
    float inv_l = 1.f / l_run;
    float ssq = 0.f;
#pragma unroll
    for (int s = 0; s < 4; s++)
#pragma unroll
        for (int r = 0; r < 8; r++) {
            oacc[s][r] *= inv_l;
            ssq += oacc[s][r] * oacc[s][r];
        }
    ssq += swap16f(ssq);
    float inv_s = 1.f / (ssq + 1e-8f);

    float* orow = att + (size_t)(qbase + nlo) * DH;
#pragma unroll
    for (int s = 0; s < 4; s++) {
        v4f w0, w1;
#pragma unroll
        for (int r = 0; r < 4; r++) {
            w0[r] = oacc[s][r]     * oacc[s][r]     * inv_s;
            w1[r] = oacc[s][4 + r] * oacc[s][4 + r] * inv_s;
        }
        *(v4f*)(orow + s * 16 + hi * 8)     = w0;
        *(v4f*)(orow + s * 16 + hi * 8 + 4) = w1;
    }
}

// ---------------------------------------------------------------------------
// Kernel 3: final rank-4 TT-linear, fp32 output
// ---------------------------------------------------------------------------
__global__ __launch_bounds__(256) void qia_out_kernel(
    const float* __restrict__ att,
    const float* __restrict__ oA, const float* __restrict__ oB, const float* __restrict__ ob,
    float* __restrict__ out)
{
    __shared__ float xs[DIM];
    __shared__ float red[8 * 4];
    __shared__ float tv[4];
    const int row = blockIdx.x;
    const int tid = threadIdx.x;
    xs[tid]       = att[(size_t)row * DIM + tid];
    xs[tid + 256] = att[(size_t)row * DIM + tid + 256];
    __syncthreads();

    float acc[4] = {0.f, 0.f, 0.f, 0.f};
    for (int d = tid; d < DIM; d += 256) {
        float xv = xs[d];
#pragma unroll
        for (int r = 0; r < 4; r++) acc[r] += xv * oB[r * DIM + d];
    }
#pragma unroll
    for (int i = 0; i < 4; i++) {
        float v = acc[i];
#pragma unroll
        for (int off = 16; off; off >>= 1) v += __shfl_xor(v, off, 32);
        if ((tid & 31) == 0) red[(tid >> 5) * 4 + i] = v;
    }
    __syncthreads();
    if (tid < 4) {
        float s = 0.f;
#pragma unroll
        for (int w = 0; w < 8; w++) s += red[w * 4 + tid];
        tv[tid] = s;
    }
    __syncthreads();

    for (int o = tid; o < DIM; o += 256) {
        float y = ob[o];
#pragma unroll
        for (int r = 0; r < 4; r++) y += tv[r] * oA[o * 4 + r];
        out[(size_t)row * DIM + o] = y;
    }
}

// ---------------------------------------------------------------------------
extern "C" void kernel_launch(void* const* d_in, const int* in_sizes, int n_in,
                              void* d_out, int out_size, void* d_ws, size_t ws_size,
                              hipStream_t stream)
{
    const float* x  = (const float*)d_in[0];
    const float* qA = (const float*)d_in[1];
    const float* qB = (const float*)d_in[2];
    const float* qb = (const float*)d_in[3];
    const float* kA = (const float*)d_in[4];
    const float* kB = (const float*)d_in[5];
    const float* kb = (const float*)d_in[6];
    const float* vA = (const float*)d_in[7];
    const float* vB = (const float*)d_in[8];
    const float* vb = (const float*)d_in[9];
    const float* oA = (const float*)d_in[10];
    const float* oB = (const float*)d_in[11];
    const float* ob = (const float*)d_in[12];

    // workspace carve-up: qh/kh/vT (f16, 2MB each) + att (f32, 4MB)
    _Float16* qh = (_Float16*)d_ws;
    _Float16* kh = qh + (size_t)NHEAD * DH;
    _Float16* vT = kh + (size_t)NHEAD * DH;
    float*    att = (float*)(vT + (size_t)NHEAD * DH);

    qia_proj_kernel<<<NROWS, 256, 0, stream>>>(x, qA, qB, qb, kA, kB, kb,
                                               vA, vB, vb, qh, kh, vT);
    qia_attn_kernel<<<(NHEAD / 16) / 8, 256, 0, stream>>>(qh, kh, vT, att);
    qia_out_kernel<<<NROWS, 256, 0, stream>>>(att, oA, oB, ob, (float*)d_out);
}
